// CentralCritic_14147622273530
// MI455X (gfx1250) — compile-verified, matrix operand order checked by reading, not run on hardware
//
#include <hip/hip_runtime.h>
#include <hip/hip_bf16.h>

// ---------------------------------------------------------------------------
// MI455X (gfx1250) implementation.
// Precision: bf16 WMMA (v_wmma_f32_16x16x32_bf16) with f32 accumulate.
// Roofline: ~34 GFLOP vs ~61 MB traffic -> compute bound; bf16 matrix path
// is ~8x the f32-WMMA (16x16x4) instruction rate, HBM floor ~2.6us.
// ---------------------------------------------------------------------------

typedef __bf16 v16bf __attribute__((ext_vector_type(16)));
typedef __bf16 v8bf  __attribute__((ext_vector_type(8)));
typedef float  v8f   __attribute__((ext_vector_type(8)));

#define NA   4        // agents
#define NH   3        // policy heads
#define NHI  2        // intrinsic heads
#define NB   32768    // batch
#define NS   256      // state dim
#define NC   16       // action dim
#define ND   128      // hidden dim
#define NIN  176      // critic input dim (3*16 + 128)
#define NKP  192      // K padded to multiple of 32

// LDS strides (in __bf16 elements); all row strides keep 16B alignment.
#define XS   200      // Xt  [128][192] pad->200
#define W1S  200      // W1t [128][192] pad->200
#define W2S  136      // W2t [16][128]  pad->136
#define H1S  136      // H1  [128][128] pad->136
#define ES   264      // encoder tiles [128][256] pad->264

// CDNA5 16-bit A/B fragment (16x32 per wave32): lane = 16*hi + m,
// element e in [0,7]  -> k = k0 + 8*hi + e        (one contiguous b128)
// element e in [8,15] -> k = k0 + 16 + 8*hi + e-8 (one contiguous b128)
__device__ __forceinline__ v16bf frag_ld(const __bf16* p /* row base + k0 */, int hi) {
    v8bf lo = *(const v8bf*)(p + (hi << 3));
    v8bf hb = *(const v8bf*)(p + 16 + (hi << 3));
    return __builtin_shufflevector(lo, hb, 0, 1, 2, 3, 4, 5, 6, 7,
                                           8, 9, 10, 11, 12, 13, 14, 15);
}

// ---------------------------------------------------------------------------
// Kernel 1: enc = relu(state @ enc_W + enc_b), stored as bf16 in workspace.
// Grid: 256 WGs x 256 threads (8 waves); 128 batch rows per WG.
// ---------------------------------------------------------------------------
__global__ __launch_bounds__(256) void enc_kernel(
    const float* __restrict__ state, const float* __restrict__ Wg,
    const float* __restrict__ bg, __bf16* __restrict__ encBf)
{
    extern __shared__ __bf16 sm[];
    __bf16* St = sm;               // [128][ES] bf16 state tile (row-major K)
    __bf16* Wt = sm + 128 * ES;    // [128][ES] bf16 enc_W transposed: Wt[n][k]

    const int tid = threadIdx.x;
    const int rowBase = blockIdx.x * 128;

    for (int i = tid; i < 128 * NS; i += 256) {
        int r = i >> 8, k = i & 255;
        St[r * ES + k] = (__bf16)state[(size_t)(rowBase + r) * NS + k];
    }
    for (int i = tid; i < 128 * NS; i += 256) {
        int n = i >> 8, k = i & 255;
        Wt[n * ES + k] = (__bf16)Wg[(size_t)k * ND + n];   // transpose on fill
    }
    __syncthreads();

    const int wave = tid >> 5, lane = tid & 31;
    const int hi = lane >> 4, lm = lane & 15;
    const __bf16* aRow = St + (wave * 16 + lm) * ES;

    const v8f vzero = {0.f, 0.f, 0.f, 0.f, 0.f, 0.f, 0.f, 0.f};
    v8f acc[8];
#pragma unroll
    for (int nt = 0; nt < 8; ++nt) acc[nt] = vzero;

    for (int kt = 0; kt < 8; ++kt) {              // K = 256 = 8 x 32
        const int k0 = kt * 32;
        v16bf af = frag_ld(aRow + k0, hi);
#pragma unroll
        for (int nt = 0; nt < 8; ++nt) {
            v16bf bf = frag_ld(Wt + (nt * 16 + lm) * ES + k0, hi);
            acc[nt] = __builtin_amdgcn_wmma_f32_16x16x32_bf16(
                false, af, false, bf, (short)0, acc[nt], false, false);
        }
    }
    // D layout: VGPR r holds (M = r + 8*hi, N = 16*nt + lm)
#pragma unroll
    for (int nt = 0; nt < 8; ++nt) {
        const int n = nt * 16 + lm;
        const float bias = bg[n];
#pragma unroll
        for (int r = 0; r < 8; ++r) {
            const int row = rowBase + wave * 16 + r + 8 * hi;
            float v = acc[nt][r] + bias;
            v = fmaxf(v, 0.f);
            encBf[(size_t)row * ND + n] = (__bf16)v;
        }
    }
}

// ---------------------------------------------------------------------------
// Kernel 2: per (critic,head,agent) combo: q = relu(x@W1+b1)@W2+b2, then
// gather q[argmax(acs[a,h,b,:])].  Grid: (256 B-tiles, 20 combos) x 256 thr.
// ---------------------------------------------------------------------------
__global__ __launch_bounds__(256) void critic_kernel(
    const float* __restrict__ acs,
    const float* __restrict__ eW1, const float* __restrict__ eb1,
    const float* __restrict__ eW2, const float* __restrict__ eb2,
    const float* __restrict__ iW1, const float* __restrict__ ib1,
    const float* __restrict__ iW2, const float* __restrict__ ib2,
    const __bf16* __restrict__ encBf, float* __restrict__ out)
{
    extern __shared__ __bf16 sm[];
    __bf16* Xt  = sm;                   // [128][NKP] critic input, row-major K
    __bf16* W1t = Xt + 128 * XS;        // [128][NKP] W1 transposed: W1t[n][k]
    __bf16* W2t = W1t + 128 * W1S;      // [16][128]  W2 transposed: W2t[n][k]
    __bf16* H1  = W2t + 16 * W2S;       // [128][128] hidden activations (bf16)
    float*  Q   = (float*)(H1 + 128 * H1S);  // [128][16] final q tile (f32)

    const int tid = threadIdx.x;
    const int rowBase = blockIdx.x * 128;
    const int combo = blockIdx.y;

    int a, h;
    const float *W1g, *b1g, *W2g, *b2g;
    size_t outBase;
    if (combo < NA * NH) {              // extrinsic critic
        a = combo / NH; h = combo % NH;
        const int w = h * NA + a;
        W1g = eW1 + (size_t)w * NIN * ND; b1g = eb1 + w * ND;
        W2g = eW2 + (size_t)w * ND * NC;  b2g = eb2 + w * NC;
        outBase = (size_t)combo * NB;
    } else {                            // intrinsic critic (heads 1..2)
        const int c2 = combo - NA * NH;
        a = c2 / NHI; const int hc = c2 % NHI; h = hc + 1;
        const int w = hc * NA + a;
        W1g = iW1 + (size_t)w * NIN * ND; b1g = ib1 + w * ND;
        W2g = iW2 + (size_t)w * ND * NC;  b2g = ib2 + w * NC;
        outBase = (size_t)(NA * NH + c2) * NB;
    }

    // ---- stage Xt: [0,48) leave-one-out actions, [48,176) enc, [176,192)=0
    for (int i = tid; i < 128 * NKP; i += 256) {
        const int r = i / NKP, k = i - r * NKP;
        const int b = rowBase + r;
        __bf16 v = (__bf16)0.f;
        if (k < 48) {
            const int t = k >> 4, c = k & 15;
            const int ka = (t < a) ? t : t + 1;        // LOO[a][t]
            v = (__bf16)acs[((size_t)(ka * NH + h) * NB + b) * NC + c];
        } else if (k < NIN) {
            v = encBf[(size_t)b * ND + (k - 48)];
        }
        Xt[r * XS + k] = v;
    }
    // ---- stage W1 transposed (bf16), zero-pad K
    for (int i = tid; i < 128 * NKP; i += 256) {
        const int n = i / NKP, k = i - n * NKP;
        W1t[n * W1S + k] = (k < NIN) ? (__bf16)W1g[(size_t)k * ND + n] : (__bf16)0.f;
    }
    // ---- stage W2 transposed (bf16)
    for (int i = tid; i < 16 * ND; i += 256) {
        const int n = i >> 7, k = i & 127;
        W2t[n * W2S + k] = (__bf16)W2g[(size_t)k * NC + n];
    }
    __syncthreads();

    const int wave = tid >> 5, lane = tid & 31;
    const int hi = lane >> 4, lm = lane & 15;
    const v8f vzero = {0.f, 0.f, 0.f, 0.f, 0.f, 0.f, 0.f, 0.f};

    // ---- GEMM1: [16 x 192] x [192 x 128] per wave, 6 k-steps x 8 n-tiles
    const __bf16* aRow = Xt + (wave * 16 + lm) * XS;
    v8f acc[8];
#pragma unroll
    for (int nt = 0; nt < 8; ++nt) acc[nt] = vzero;
    for (int kt = 0; kt < 6; ++kt) {
        const int k0 = kt * 32;
        v16bf af = frag_ld(aRow + k0, hi);
#pragma unroll
        for (int nt = 0; nt < 8; ++nt) {
            v16bf bf = frag_ld(W1t + (nt * 16 + lm) * W1S + k0, hi);
            acc[nt] = __builtin_amdgcn_wmma_f32_16x16x32_bf16(
                false, af, false, bf, (short)0, acc[nt], false, false);
        }
    }
    // bias + relu -> H1 (rows owned exclusively by this wave; no barrier)
#pragma unroll
    for (int nt = 0; nt < 8; ++nt) {
        const int n = nt * 16 + lm;
        const float bias = b1g[n];
#pragma unroll
        for (int r = 0; r < 8; ++r) {
            const int mrow = wave * 16 + r + 8 * hi;
            float v = acc[nt][r] + bias;
            H1[mrow * H1S + n] = (__bf16)fmaxf(v, 0.f);
        }
    }

    // ---- GEMM2: [16 x 128] x [128 x 16] per wave, 4 k-steps, one n-tile
    v8f acc2 = vzero;
    const __bf16* a2Row = H1 + (wave * 16 + lm) * H1S;
    for (int kt = 0; kt < 4; ++kt) {
        const int k0 = kt * 32;
        v16bf af = frag_ld(a2Row + k0, hi);
        v16bf bf = frag_ld(W2t + lm * W2S + k0, hi);
        acc2 = __builtin_amdgcn_wmma_f32_16x16x32_bf16(
            false, af, false, bf, (short)0, acc2, false, false);
    }
    const float bias2 = b2g[lm];
#pragma unroll
    for (int r = 0; r < 8; ++r) {
        const int mrow = wave * 16 + r + 8 * hi;
        Q[mrow * 16 + lm] = acc2[r] + bias2;
    }
    __syncthreads();

    // ---- argmax over agent's own actions (first max, like jnp.argmax) + gather
    if (tid < 128) {
        const int b = rowBase + tid;
        const float* p = acs + ((size_t)(a * NH + h) * NB + b) * NC;
        int best = 0;
        float bv = p[0];
#pragma unroll
        for (int c = 1; c < NC; ++c) {
            const float v = p[c];
            if (v > bv) { bv = v; best = c; }
        }
        out[outBase + b] = Q[tid * 16 + best];
    }
}

// ---------------------------------------------------------------------------
extern "C" void kernel_launch(void* const* d_in, const int* in_sizes, int n_in,
                              void* d_out, int out_size, void* d_ws, size_t ws_size,
                              hipStream_t stream) {
    const float* state = (const float*)d_in[0];
    const float* acs   = (const float*)d_in[1];
    const float* enc_W = (const float*)d_in[2];
    const float* enc_b = (const float*)d_in[3];
    const float* eW1   = (const float*)d_in[4];
    const float* eb1   = (const float*)d_in[5];
    const float* eW2   = (const float*)d_in[6];
    const float* eb2   = (const float*)d_in[7];
    const float* iW1   = (const float*)d_in[8];
    const float* ib1   = (const float*)d_in[9];
    const float* iW2   = (const float*)d_in[10];
    const float* ib2   = (const float*)d_in[11];

    __bf16* encBf = (__bf16*)d_ws;   // B*D bf16 = 8 MB scratch

    constexpr size_t ENC_LDS  = (size_t)2 * 128 * ES * sizeof(__bf16);          // 135168
    constexpr size_t CRIT_LDS = ((size_t)128 * XS + 128 * W1S + 16 * W2S +
                                 128 * H1S) * sizeof(__bf16) +
                                (size_t)128 * 16 * sizeof(float);               // 149760

    enc_kernel<<<dim3(NB / 128), dim3(256), ENC_LDS, stream>>>(
        state, enc_W, enc_b, encBf);

    critic_kernel<<<dim3(NB / 128, NA * NH + NA * NHI), dim3(256), CRIT_LDS, stream>>>(
        acs, eW1, eb1, eW2, eb2, iW1, ib1, iW2, ib2, encBf, (float*)d_out);
}
